// PAMNet_83863531422359
// MI455X (gfx1250) — matrix operand order repaired and checked by reading, not compiled, not run on hardware
//
// PAMNet forward for MI455X (gfx1250, wave32, WMMA).
// All 128-wide GEMMs run through v_wmma_f32_16x16x32_bf16 with f32 accum;
// graph gathers/scatters (segment_sum, gathered-mul) are fused into the GEMM
// prologue/epilogue (AI ~43 FLOP/B vs 23.3 TB/s HBM -> compute-bound).
// 128-wide LDS K-tiles (52KB) -> one barrier pair per K=128, 16 WMMAs
// back-to-back per wave. GEMM specialized by template to strip runtime
// branches from staging/epilogue.
#include <hip/hip_runtime.h>
#include <math.h>

#define NMOL 256
#define TILE_M 64
#define TILE_N 128
#define KTILE 128
#define KPAD 136   // +8 u16 padding against LDS bank conflicts

typedef __attribute__((ext_vector_type(16))) __bf16 v16bf;
typedef __attribute__((ext_vector_type(8)))  float  v8f;
union Frag { v16bf v; unsigned short u[16]; };

__device__ __forceinline__ unsigned short f2bf(float f) {
  unsigned int x = __float_as_uint(f);
  unsigned int r = x + 0x7FFFu + ((x >> 16) & 1u);   // round-to-nearest-even
  return (unsigned short)(r >> 16);
}
__device__ __forceinline__ float siluf(float v) { return v / (1.f + __expf(-v)); }
__device__ __forceinline__ float envf(float x) { // P_ENV=6 -> a=-28,b=48,c=-21
  if (x >= 1.f) return 0.f;
  float x2 = x * x, x5 = x2 * x2 * x;
  return 1.f / x - 28.f * x5 + 48.f * x5 * x - 21.f * x5 * x2;
}
__device__ __forceinline__ float fjn(int l, float x) {
  float s, c; __sincosf(x, &s, &c);
  float j0 = s / x;
  if (l == 0) return j0;
  float jm1 = j0, jc = s / (x * x) - c / x;
  for (int ll = 1; ll < l; ++ll) { float t = (2 * ll + 1) / x * jc - jm1; jm1 = jc; jc = t; }
  return jc;
}
__device__ double djn(int l, double x) {
  double j0 = sin(x) / x;
  if (l == 0) return j0;
  double jm1 = j0, jc = sin(x) / (x * x) - cos(x) / x;
  for (int ll = 1; ll < l; ++ll) { double t = (2 * ll + 1) / x * jc - jm1; jm1 = jc; jc = t; }
  return jc;
}

// ---------------------------------------------------------------------------
// Fused WMMA GEMM:  Y = epilogue( act( A' @ W^T + bias ) )
//   GATHER: A' row = [A[gi[row]], A[gj[row]], gea[row]] (edge concat, K=384)
//   ACT:    1 = SiLU
//   HASADD: += addend[row] after activation (residual)
//   MULM:   1 = *mulg[row], 2 = *mulg[mulidx[row]] (gathered multiply)
//   SCAT:   atomicAdd into Y[scat[row]] (fused segment_sum) else store Y[row]
// ---------------------------------------------------------------------------
template <bool GATHER, int ACT, bool HASADD, int MULM, bool SCAT>
__global__ void __launch_bounds__(256)
pamnet_wmma_gemm(const float* __restrict__ A, int lda, int aoff,
                 const int* __restrict__ gi, const int* __restrict__ gj,
                 const float* __restrict__ gea,
                 const float* __restrict__ W, const float* __restrict__ bias,
                 const float* __restrict__ addend,
                 const float* __restrict__ mulg, const int* __restrict__ mulidx,
                 float* __restrict__ Y, const int* __restrict__ scat,
                 int M, int K, int N) {
  __shared__ unsigned short sA[TILE_M][KPAD];
  __shared__ unsigned short sB[TILE_N][KPAD];
  const int tid = threadIdx.x;
  const int wave = tid >> 5, lane = tid & 31;
  const int wm = wave >> 1, wn = wave & 1;
  const int half = lane >> 4, l16 = lane & 15;
  const int rowbase = blockIdx.x * TILE_M;
  const int colbase = blockIdx.y * TILE_N;

  v8f acc[4];
#pragma unroll
  for (int nb = 0; nb < 4; ++nb) acc[nb] = (v8f)0.0f;

  for (int k0 = 0; k0 < K; k0 += KTILE) {
    // stage A tile: 64 rows x 64 bf16-pairs, packed b32 stores to LDS
    for (int t = tid; t < TILE_M * (KTILE / 2); t += 256) {
      int r = t >> 6, kp = t & 63;
      int grow = rowbase + r, gk = k0 + 2 * kp;
      float v0 = 0.f, v1 = 0.f;
      if (grow < M) {
        if (GATHER) {
          int ri = gi[grow], rj = gj[grow];
          if (gk < K)
            v0 = (gk < 128) ? A[(size_t)ri * 128 + gk]
               : (gk < 256) ? A[(size_t)rj * 128 + (gk - 128)]
                            : gea[(size_t)grow * 128 + (gk - 256)];
          if (gk + 1 < K)
            v1 = (gk + 1 < 128) ? A[(size_t)ri * 128 + gk + 1]
               : (gk + 1 < 256) ? A[(size_t)rj * 128 + (gk - 127)]
                                : gea[(size_t)grow * 128 + (gk - 255)];
        } else {
          if (gk < K)     v0 = A[(size_t)grow * lda + aoff + gk];
          if (gk + 1 < K) v1 = A[(size_t)grow * lda + aoff + gk + 1];
        }
      }
      *(unsigned*)&sA[r][2 * kp] = (unsigned)f2bf(v0) | ((unsigned)f2bf(v1) << 16);
    }
    // stage B tile: 128 rows x 64 bf16-pairs
    for (int t = tid; t < TILE_N * (KTILE / 2); t += 256) {
      int nn = t >> 6, kp = t & 63;
      int gn = colbase + nn, gk = k0 + 2 * kp;
      float v0 = 0.f, v1 = 0.f;
      if (gn < N) {
        if (gk < K)     v0 = W[(size_t)gn * K + gk];
        if (gk + 1 < K) v1 = W[(size_t)gn * K + gk + 1];
        if (gk + KTILE < K) __builtin_prefetch(&W[(size_t)gn * K + gk + KTILE], 0, 0);
      }
      *(unsigned*)&sB[nn][2 * kp] = (unsigned)f2bf(v0) | ((unsigned)f2bf(v1) << 16);
    }
    __syncthreads();

#pragma unroll
    for (int ks = 0; ks < KTILE; ks += 32) {   // 16 WMMAs per wave per tile
      Frag a;  // A 16x32 bf16 (ISA 7.12.2: lane m=L&15, K half by lane>=16)
#pragma unroll
      for (int i = 0; i < 16; ++i) {
        int vg = i >> 1, lb = i & 1;
        int kk = ((vg < 4) ? (2 * vg) : (16 + 2 * (vg - 4))) + 8 * half + lb;
        a.u[i] = sA[wm * 16 + l16][ks + kk];
      }
#pragma unroll
      for (int nb = 0; nb < 4; ++nb) {
        Frag b;  // B 32x16 bf16: col n=L&15, K = 8*half + (i&7) + 16*(i>>3), same as A
#pragma unroll
        for (int i = 0; i < 16; ++i)
          b.u[i] = sB[wn * 64 + nb * 16 + l16][ks + (i & 7) + 8 * half + 16 * (i >> 3)];
        acc[nb] = __builtin_amdgcn_wmma_f32_16x16x32_bf16(
            false, a.v, false, b.v, (short)0, acc[nb], false, false);
      }
    }
    __syncthreads();
  }

#pragma unroll
  for (int nb = 0; nb < 4; ++nb) {
#pragma unroll
    for (int r = 0; r < 8; ++r) {               // C/D: M = r + 8*half, N = l16
      int grow = rowbase + wm * 16 + r + 8 * half;
      int gcol = colbase + wn * 64 + nb * 16 + l16;
      if (grow < M && gcol < N) {
        float v = acc[nb][r];
        if (bias) v += bias[gcol];
        if (ACT == 1) v = siluf(v);
        if (HASADD) v += addend[(size_t)grow * N + gcol];
        if (MULM == 1) v *= mulg[(size_t)grow * N + gcol];
        if (MULM == 2) v *= mulg[(size_t)mulidx[grow] * N + gcol];
        if (SCAT) atomicAdd(&Y[(size_t)scat[grow] * N + gcol], v);
        else      Y[(size_t)grow * N + gcol] = v;
      }
    }
  }
}

// --------------------------- small helper kernels ---------------------------
__global__ void pamnet_zero(float* p, size_t n) {
  size_t i = (size_t)blockIdx.x * blockDim.x + threadIdx.x;
  if (i < n) p[i] = 0.f;
}
__global__ void pamnet_addto(float* dst, const float* src, size_t n) {
  size_t i = (size_t)blockIdx.x * blockDim.x + threadIdx.x;
  if (i < n) dst[i] += src[i];
}
__global__ void pamnet_bessel_tables(float* zTab, float* nTab) {
  if (threadIdx.x || blockIdx.x) return;
  double zeros[7][13];
  for (int k = 0; k < 13; ++k) zeros[0][k] = (k + 1) * 3.14159265358979323846;
  for (int l = 1; l < 7; ++l)
    for (int k = 0; k < 13 - l; ++k) {
      double a = zeros[l - 1][k], b = zeros[l - 1][k + 1];
      double fa = djn(l, a);
      for (int it = 0; it < 80; ++it) {
        double m = 0.5 * (a + b), fm = djn(l, m);
        if (fa * fm <= 0) b = m; else { a = m; fa = fm; }
      }
      zeros[l][k] = 0.5 * (a + b);
    }
  for (int l = 0; l < 7; ++l)
    for (int r = 0; r < 6; ++r) {
      zTab[l * 6 + r] = (float)zeros[l][r];
      nTab[l * 6 + r] = (float)(sqrt(2.0) / fabs(djn(l + 1, zeros[l][r])));
    }
}
__global__ void pamnet_edge_rbf(const float* __restrict__ xraw,
                                const int* __restrict__ jrow, const int* __restrict__ icol,
                                const float* __restrict__ freq,
                                float* __restrict__ dist, float* __restrict__ rbf,
                                int E, float cutoff, float dmin) {
  int e = blockIdx.x * blockDim.x + threadIdx.x;
  if (e >= E) return;
  int j = jrow[e], i = icol[e];
  float dx = xraw[(size_t)i * 22 + 0] - xraw[(size_t)j * 22 + 0];
  float dy = xraw[(size_t)i * 22 + 1] - xraw[(size_t)j * 22 + 1];
  float dz = xraw[(size_t)i * 22 + 2] - xraw[(size_t)j * 22 + 2];
  float d = fmaxf(sqrtf(dx * dx + dy * dy + dz * dz), dmin);
  dist[e] = d;
  float x = d / cutoff, env = envf(x);
  for (int f = 0; f < 16; ++f) rbf[(size_t)e * 16 + f] = env * __sinf(freq[f] * x);
}
__global__ void pamnet_triplet_angle(const float* __restrict__ xraw,
                                     const int* __restrict__ ia, const int* __restrict__ ib,
                                     const int* __restrict__ ic, float* __restrict__ ang, int T) {
  int t = blockIdx.x * blockDim.x + threadIdx.x;
  if (t >= T) return;
  int a = ia[t], b = ib[t], c = ic[t];
  float v1x = xraw[(size_t)b * 22 + 0] - xraw[(size_t)a * 22 + 0];
  float v1y = xraw[(size_t)b * 22 + 1] - xraw[(size_t)a * 22 + 1];
  float v1z = xraw[(size_t)b * 22 + 2] - xraw[(size_t)a * 22 + 2];
  float v2x = xraw[(size_t)c * 22 + 0] - xraw[(size_t)b * 22 + 0];
  float v2y = xraw[(size_t)c * 22 + 1] - xraw[(size_t)b * 22 + 1];
  float v2z = xraw[(size_t)c * 22 + 2] - xraw[(size_t)b * 22 + 2];
  float cx = v1y * v2z - v1z * v2y, cy = v1z * v2x - v1x * v2z, cz = v1x * v2y - v1y * v2x;
  ang[t] = atan2f(sqrtf(cx * cx + cy * cy + cz * cz), v1x * v2x + v1y * v2y + v1z * v2z);
}
__global__ void pamnet_sbf_radial(const float* __restrict__ dist,
                                  const float* __restrict__ zTab, const float* __restrict__ nTab,
                                  float* __restrict__ rad, int E, float cutoff) {
  int e = blockIdx.x * blockDim.x + threadIdx.x;
  if (e >= E) return;
  float x = dist[e] / cutoff, env = envf(x);
  for (int l = 0; l < 7; ++l)
    for (int r = 0; r < 6; ++r)
      rad[(size_t)e * 42 + l * 6 + r] = env * nTab[l * 6 + r] * fjn(l, zTab[l * 6 + r] * x);
}
__global__ void pamnet_sbf_combine(const float* __restrict__ rad, const float* __restrict__ ang,
                                   const int* __restrict__ idx_e, float* __restrict__ out, int T) {
  int t = blockIdx.x * blockDim.x + threadIdx.x;
  if (t >= T) return;
  float ca = __cosf(ang[t]);
  float P[7]; P[0] = 1.f; P[1] = ca;
  for (int l = 1; l <= 5; ++l) P[l + 1] = ((2 * l + 1) * ca * P[l] - l * P[l - 1]) / (l + 1);
  int e = idx_e[t];
  const float inv4pi = 0.0795774715459476f;
  for (int l = 0; l < 7; ++l) {
    float yc = sqrtf((2 * l + 1) * inv4pi) * P[l];
    for (int r = 0; r < 6; ++r)
      out[(size_t)t * 42 + l * 6 + r] = rad[(size_t)e * 42 + l * 6 + r] * yc;
  }
}
__global__ void pamnet_proj1(const float* __restrict__ X, const float* __restrict__ w,
                             const float* __restrict__ b, float* __restrict__ y, int M) {
  int r = blockIdx.x * blockDim.x + threadIdx.x;
  if (r >= M) return;
  float acc = 0.f;
  for (int c = 0; c < 128; ++c) acc += X[(size_t)r * 128 + c] * w[c];
  y[r] = acc + b[0];
}
__global__ void pamnet_att_reduce(const float* __restrict__ og, const float* __restrict__ ag,
                                  const float* __restrict__ ol, const float* __restrict__ al,
                                  float* __restrict__ s, int total) {
  __shared__ float red[256];
  int r = blockIdx.x * 256 + threadIdx.x;
  float c = 0.f;
  if (r < total) {
    float a0 = ag[r], a1 = al[r];
    a0 = a0 > 0.f ? a0 : 0.2f * a0;
    a1 = a1 > 0.f ? a1 : 0.2f * a1;
    float mx = fmaxf(a0, a1);
    float e0 = __expf(a0 - mx), e1 = __expf(a1 - mx);
    float inv = 1.f / (e0 + e1);
    c = og[r] * e0 * inv + ol[r] * e1 * inv;
  }
  red[threadIdx.x] = c; __syncthreads();
  for (int o = 128; o; o >>= 1) { if (threadIdx.x < o) red[threadIdx.x] += red[threadIdx.x + o]; __syncthreads(); }
  if (!threadIdx.x) atomicAdd(s, red[0]);
}
__global__ void pamnet_final_out(const float* __restrict__ xraw, const int* __restrict__ batch,
                                 const float* __restrict__ s, float* __restrict__ out, int n) {
  int v = blockIdx.x * blockDim.x + threadIdx.x;
  if (v >= n) return;
  float ai = xraw[(size_t)v * 22 + 0] > 0.f ? -1.f : 1.f;
  atomicAdd(&out[batch[v]], s[0] * ai);
}

// --------------------------------- driver -----------------------------------
struct TailP { const float *r1W1,*r1B1,*r1W2,*r1B2,*r2W1,*r2B1,*r2W2,*r2B2,
                          *r3W1,*r3B1,*r3W2,*r3B2,*h2W,*h2B,
                          *yW1,*yB1,*yW2,*yB2,*outW,*outB,*attW,*attB; };
struct GlobP { const float *h1W,*h1B,*mW,*mB; TailP t; };
struct LocP  { const float *h1W,*h1B,*mW,*mB,*kjW,*kjB,*jjW,*jjB,*jiW,*jiB,
                          *Wsbf1,*Wsbf2,*Wrbf; TailP t; };

extern "C" void kernel_launch(void* const* d_in, const int* in_sizes, int n_in,
                              void* d_out, int out_size, void* d_ws, size_t ws_size,
                              hipStream_t stream) {
  (void)out_size; (void)ws_size;
  const float* xraw = (const float*)d_in[0];
  const int n = in_sizes[0] / 22;

  // Params: JAX pytree leaves in _make_params insertion order. Supports either
  // one flat f32 blob at d_in[1] (n_in==15) or one d_in slot per leaf.
  struct PSrc {
    void* const* din; int slot; const float* blob; size_t off; bool exp;
    const float* take(size_t nf) {
      if (exp) return (const float*)din[slot++];
      const float* p = blob + off; off += nf; return p;
    }
  } P;
  P.din = d_in; P.slot = 1; P.blob = (const float*)d_in[1]; P.off = 0;
  P.exp = (n_in > 30);

  const float* initW  = P.take(128 * 18);
  const float* freq_g = P.take(16);
  const float* freq_l = P.take(16);
  const float* rbfgW = P.take(128 * 16); const float* rbfgB = P.take(128);
  const float* rbflW = P.take(128 * 16); const float* rbflB = P.take(128);
  const float* sbf1W = P.take(128 * 42); const float* sbf1B = P.take(128);
  const float* sbf2W = P.take(128 * 42); const float* sbf2B = P.take(128);
  GlobP gp[3]; LocP lp[3];
  auto takeTail = [&](TailP& t) {
    t.r1W1=P.take(128*128); t.r1B1=P.take(128); t.r1W2=P.take(128*128); t.r1B2=P.take(128);
    t.r2W1=P.take(128*128); t.r2B1=P.take(128); t.r2W2=P.take(128*128); t.r2B2=P.take(128);
    t.r3W1=P.take(128*128); t.r3B1=P.take(128); t.r3W2=P.take(128*128); t.r3B2=P.take(128);
  };
  for (int L = 0; L < 3; ++L) {
    GlobP& g = gp[L];
    g.h1W=P.take(128*128); g.h1B=P.take(128); g.mW=P.take(128*384); g.mB=P.take(128);
    takeTail(g.t);
    g.t.h2W=P.take(128*128); g.t.h2B=P.take(128);
    g.t.yW1=P.take(128*128); g.t.yB1=P.take(128); g.t.yW2=P.take(128*128); g.t.yB2=P.take(128);
    g.t.outW=P.take(128); g.t.outB=P.take(1); g.t.attW=P.take(128); g.t.attB=P.take(1);
  }
  for (int L = 0; L < 3; ++L) {
    LocP& l = lp[L];
    l.h1W=P.take(128*128); l.h1B=P.take(128); l.mW=P.take(128*384); l.mB=P.take(128);
    l.kjW=P.take(128*128); l.kjB=P.take(128); l.jjW=P.take(128*128); l.jjB=P.take(128);
    l.jiW=P.take(128*128); l.jiB=P.take(128);
    l.Wsbf1=P.take(128*128); l.Wsbf2=P.take(128*128); l.Wrbf=P.take(128*128);
    takeTail(l.t);
    l.t.h2W=P.take(128*128); l.t.h2B=P.take(128);
    l.t.yW1=P.take(128*128); l.t.yB1=P.take(128); l.t.yW2=P.take(128*128); l.t.yB2=P.take(128);
    l.t.outW=P.take(128); l.t.outB=P.take(1); l.t.attW=P.take(128); l.t.attB=P.take(1);
  }
  const int base = P.exp ? P.slot : 2;
  const int* batch    = (const int*)d_in[base + 0];
  const int* eg       = (const int*)d_in[base + 1];
  const int* el       = (const int*)d_in[base + 2];
  const int* idx_i    = (const int*)d_in[base + 3];
  const int* idx_j    = (const int*)d_in[base + 4];
  const int* idx_k    = (const int*)d_in[base + 5];
  const int* idx_kj   = (const int*)d_in[base + 6];
  const int* idx_ji   = (const int*)d_in[base + 7];
  const int* idx_ip   = (const int*)d_in[base + 8];
  const int* idx_j1p  = (const int*)d_in[base + 9];
  const int* idx_j2p  = (const int*)d_in[base + 10];
  const int* idx_jjp  = (const int*)d_in[base + 11];
  const int* idx_jip  = (const int*)d_in[base + 12];
  const int Eg = in_sizes[base + 1] / 2, El = in_sizes[base + 2] / 2;
  const int T2 = in_sizes[base + 6],     T1 = in_sizes[base + 11];
  const int* jg = eg; const int* ig = eg + Eg;
  const int* jl = el; const int* il = el + El;

  // Workspace bump allocator (256B aligned).
  char* wsp = (char*)d_ws;
  auto alloc = [&](size_t nf) {
    float* p = (float*)wsp; wsp += ((nf * sizeof(float)) + 255) & ~(size_t)255; return p;
  };
  float* zTab = alloc(42);        float* nTab = alloc(42);
  float* dist_g = alloc(Eg);      float* rbf_g = alloc((size_t)Eg * 16);
  float* dist_l = alloc(El);      float* rbf_l = alloc((size_t)El * 16);
  float* ang2 = alloc(T2);        float* ang1 = alloc(T1);
  float* rad  = alloc((size_t)El * 42);
  float* sbf2b = alloc((size_t)T2 * 42);  float* sbf1b = alloc((size_t)T1 * 42);
  float* ea_g  = alloc((size_t)Eg * 128); float* ea_l  = alloc((size_t)El * 128);
  float* ea_s1 = alloc((size_t)T1 * 128); float* ea_s2 = alloc((size_t)T2 * 128);
  float* xA = alloc((size_t)n * 128);  float* xB = alloc((size_t)n * 128);
  float* nb1 = alloc((size_t)n * 128); float* nb2 = alloc((size_t)n * 128);
  float* nb3 = alloc((size_t)n * 128); float* nb4 = alloc((size_t)n * 128);
  float* mbuf = alloc((size_t)El * 128); float* eTmp = alloc((size_t)El * 128);
  float* macc = alloc((size_t)El * 128);
  float* og3 = alloc(3 * (size_t)n); float* ag3 = alloc(3 * (size_t)n);
  float* ol3 = alloc(3 * (size_t)n); float* al3 = alloc(3 * (size_t)n);
  float* sS  = alloc(1);

  auto zero = [&](float* p, size_t nf) {
    pamnet_zero<<<dim3((unsigned)((nf + 255) / 256)), dim3(256), 0, stream>>>(p, nf);
  };
  // Compile-time specialized dispatch over epilogue/prologue fusion modes.
  auto gemm = [&](const float* A, int lda, int aoff,
                  const int* gi, const int* gj, const float* gea,
                  const float* W, const float* bias, const float* addend,
                  const float* mulg, const int* mulidx,
                  float* Y, const int* scat, int M, int K, int N, int act) {
    dim3 grid((M + TILE_M - 1) / TILE_M, (N + TILE_N - 1) / TILE_N), blk(256);
    const bool G = gi != nullptr, D = addend != nullptr, S = scat != nullptr;
    const int  MU = mulg ? (mulidx ? 2 : 1) : 0;
#define PAMNET_GO(g_, a_, d_, m_, s_) \
    pamnet_wmma_gemm<g_, a_, d_, m_, s_><<<grid, blk, 0, stream>>>( \
        A, lda, aoff, gi, gj, gea, W, bias, addend, mulg, mulidx, Y, scat, M, K, N)
    if      (!G && act == 0 && !D && MU == 0 && !S) PAMNET_GO(false, 0, false, 0, false);
    else if (!G && act == 1 && !D && MU == 0 && !S) PAMNET_GO(false, 1, false, 0, false);
    else if (!G && act == 1 &&  D && MU == 0 && !S) PAMNET_GO(false, 1, true,  0, false);
    else if ( G && act == 1 && !D && MU == 0 && !S) PAMNET_GO(true,  1, false, 0, false);
    else if ( G && act == 1 && !D && MU == 0 &&  S) PAMNET_GO(true,  1, false, 0, true);
    else if (!G && act == 0 && !D && MU == 2 &&  S) PAMNET_GO(false, 0, false, 2, true);
    else if (!G && act == 0 && !D && MU == 1 &&  S) PAMNET_GO(false, 0, false, 1, true);
#undef PAMNET_GO
  };
  auto proj = [&](const float* X, const float* w, const float* b, float* y) {
    pamnet_proj1<<<dim3((n + 255) / 256), dim3(256), 0, stream>>>(X, w, b, y, n);
  };
  auto tail = [&](const TailP& t, float* hsum, const float* resx,
                  float* xout, float* ogD, float* agD) {
    gemm(hsum,128,0, 0,0,0, t.r1W1,t.r1B1, 0, 0,0, nb3, 0, n,128,128, 1);
    gemm(nb3, 128,0, 0,0,0, t.r1W2,t.r1B2, hsum, 0,0, nb4, 0, n,128,128, 1);
    gemm(nb4, 128,0, 0,0,0, t.h2W, t.h2B,  resx, 0,0, nb3, 0, n,128,128, 1);
    gemm(nb3, 128,0, 0,0,0, t.r2W1,t.r2B1, 0, 0,0, nb1, 0, n,128,128, 1);
    gemm(nb1, 128,0, 0,0,0, t.r2W2,t.r2B2, nb3, 0,0, nb4, 0, n,128,128, 1);
    gemm(nb4, 128,0, 0,0,0, t.r3W1,t.r3B1, 0, 0,0, nb1, 0, n,128,128, 1);
    gemm(nb1, 128,0, 0,0,0, t.r3W2,t.r3B2, nb4, 0,0, xout, 0, n,128,128, 1);
    gemm(xout,128,0, 0,0,0, t.yW1, t.yB1,  0, 0,0, nb1, 0, n,128,128, 1);
    gemm(nb1, 128,0, 0,0,0, t.yW2, t.yB2,  0, 0,0, nb3, 0, n,128,128, 1);
    proj(nb3, t.outW, t.outB, ogD); proj(nb3, t.attW, t.attB, agD);
  };

  // ---- basis functions ----
  pamnet_bessel_tables<<<dim3(1), dim3(1), 0, stream>>>(zTab, nTab);
  pamnet_edge_rbf<<<dim3((Eg + 255) / 256), dim3(256), 0, stream>>>(xraw, jg, ig, freq_g, dist_g, rbf_g, Eg, 5.0f, 0.1f);
  pamnet_edge_rbf<<<dim3((El + 255) / 256), dim3(256), 0, stream>>>(xraw, jl, il, freq_l, dist_l, rbf_l, El, 3.0f, 0.01f);
  pamnet_triplet_angle<<<dim3((T2 + 255) / 256), dim3(256), 0, stream>>>(xraw, idx_i, idx_j, idx_k, ang2, T2);
  pamnet_triplet_angle<<<dim3((T1 + 255) / 256), dim3(256), 0, stream>>>(xraw, idx_ip, idx_j1p, idx_j2p, ang1, T1);
  pamnet_sbf_radial<<<dim3((El + 255) / 256), dim3(256), 0, stream>>>(dist_l, zTab, nTab, rad, El, 3.0f);
  pamnet_sbf_combine<<<dim3((T2 + 255) / 256), dim3(256), 0, stream>>>(rad, ang2, idx_kj, sbf2b, T2);
  pamnet_sbf_combine<<<dim3((T1 + 255) / 256), dim3(256), 0, stream>>>(rad, ang1, idx_jjp, sbf1b, T1);

  // ---- edge-attribute MLPs (WMMA, K zero-padded in LDS) ----
  gemm(rbf_g, 16, 0, 0,0,0, rbfgW, rbfgB, 0, 0,0, ea_g,  0, Eg, 16, 128, 1);
  gemm(rbf_l, 16, 0, 0,0,0, rbflW, rbflB, 0, 0,0, ea_l,  0, El, 16, 128, 1);
  gemm(sbf1b, 42, 0, 0,0,0, sbf1W, sbf1B, 0, 0,0, ea_s1, 0, T1, 42, 128, 1);
  gemm(sbf2b, 42, 0, 0,0,0, sbf2W, sbf2B, 0, 0,0, ea_s2, 0, T2, 42, 128, 1);

  // ---- node init: x = x_raw[:,3:21] @ init_W.T ----
  gemm(xraw, 22, 3, 0,0,0, initW, 0, 0, 0,0, xA, 0, n, 18, 128, 0);

  float* xcur = xA; float* xnxt = xB;
  for (int L = 0; L < 3; ++L) {
    // global layer: fused concat-gather A and fused segment_sum scatter
    const GlobP& g = gp[L];
    gemm(xcur, 128, 0, 0,0,0, g.h1W, g.h1B, 0, 0,0, nb1, 0, n, 128, 128, 1);
    zero(nb2, (size_t)n * 128);
    gemm(nb1, 0, 0, ig, jg, ea_g, g.mW, g.mB, 0, 0,0, nb2, ig, Eg, 384, 128, 1);
    tail(g.t, nb2, xcur, xnxt, og3 + (size_t)L * n, ag3 + (size_t)L * n);
    { float* t = xcur; xcur = xnxt; xnxt = t; }

    // local layer: triplet paths fused as GEMM + gathered-mul + atomic scatter
    const LocP& l = lp[L];
    gemm(xcur, 128, 0, 0,0,0, l.h1W, l.h1B, 0, 0,0, nb1, 0, n, 128, 128, 1);
    gemm(nb1, 0, 0, il, jl, ea_l, l.mW, l.mB, 0, 0,0, mbuf, 0, El, 384, 128, 1);
    gemm(mbuf, 128, 0, 0,0,0, l.kjW, l.kjB, 0, 0,0, eTmp, 0, El, 128, 128, 1);
    zero(macc, (size_t)El * 128);
    gemm(ea_s2, 128, 0, 0,0,0, l.Wsbf2, 0, 0, eTmp, idx_kj,  macc, idx_ji,  T2, 128, 128, 0);
    gemm(mbuf, 128, 0, 0,0,0, l.jjW, l.jjB, 0, 0,0, eTmp, 0, El, 128, 128, 1);
    gemm(ea_s1, 128, 0, 0,0,0, l.Wsbf1, 0, 0, eTmp, idx_jjp, macc, idx_jip, T1, 128, 128, 0);
    gemm(mbuf, 128, 0, 0,0,0, l.jiW, l.jiB, 0, 0,0, eTmp, 0, El, 128, 128, 1);
    pamnet_addto<<<dim3((unsigned)(((size_t)El * 128 + 255) / 256)), dim3(256), 0, stream>>>(macc, eTmp, (size_t)El * 128);
    zero(nb2, (size_t)n * 128);
    gemm(ea_l, 128, 0, 0,0,0, l.Wrbf, 0, 0, macc, 0, nb2, il, El, 128, 128, 0);
    tail(l.t, nb2, xcur, xnxt, ol3 + (size_t)L * n, al3 + (size_t)L * n);
    { float* t = xcur; xcur = xnxt; xnxt = t; }
  }

  // ---- attention softmax + global reduction ----
  zero(sS, 1);
  zero((float*)d_out, NMOL);
  pamnet_att_reduce<<<dim3((3 * n + 255) / 256), dim3(256), 0, stream>>>(og3, ag3, ol3, al3, sS, 3 * n);
  pamnet_final_out<<<dim3((n + 255) / 256), dim3(256), 0, stream>>>(xraw, batch, sS, (float*)d_out, n);
}
